// AffineChannelAttention_50809463111992
// MI455X (gfx1250) — matrix operand order repaired and checked by reading, not compiled, not run on hardware
//
#include <hip/hip_runtime.h>
#include <hip/hip_bf16.h>

// Problem constants (match reference)
#define B_ROWS   16384
#define L_LEADS  12
#define F_DIM    256
#define H_DIM    2048
#define IN_DIM   (L_LEADS * F_DIM)
#define NEG_BIG  (-1000000000.0f)

// --- WMMA vector types (CDNA5 gfx1250, wave32) ---
typedef __attribute__((ext_vector_type(16))) __bf16 v16bf;
typedef __attribute__((ext_vector_type(8)))  float  v8f;

struct Raw32B { uint4 a; uint4 b; };  // 32 bytes = one v16bf fragment

// ---- bf16 helpers (manual RNE, avoids relying on __bf16 scalar arithmetic) ----
__device__ __forceinline__ unsigned short f32_to_bf16(float f) {
  unsigned int u = __float_as_uint(f);
  u += 0x7FFFu + ((u >> 16) & 1u);   // round-to-nearest-even
  return (unsigned short)(u >> 16);
}
__device__ __forceinline__ float bf16_to_f32(unsigned short h) {
  return __uint_as_float(((unsigned int)h) << 16);
}

// A-matrix fragment (16x32 bf16): lane<16 -> row m0+lane, K {kb..kb+7, kb+16..kb+23}
// (caller passes p = rowBase + kb; chunks at p and p+16 halves)
__device__ __forceinline__ v16bf load_a_frag(const unsigned short* p) {
  Raw32B t;
  t.a = *(const uint4*)(p);
  t.b = *(const uint4*)(p + 16);
  return __builtin_bit_cast(v16bf, t);
}
// B-matrix fragment: pre-packed contiguous 32 bytes per lane
__device__ __forceinline__ v16bf load_b_frag(const unsigned short* p) {
  Raw32B t;
  t.a = *(const uint4*)(p);
  t.b = *(const uint4*)(p + 8);
  return __builtin_bit_cast(v16bf, t);
}

__device__ __forceinline__ v8f wmma_bf16(v16bf a, v16bf b, v8f c) {
  return __builtin_amdgcn_wmma_f32_16x16x32_bf16(
      /*neg_a=*/false, a, /*neg_b=*/false, b,
      /*c_mod=*/(short)0, c, /*reuse_a=*/false, /*reuse_b=*/false);
}

// =====================================================================
// Kernel 0: split W (256 x 2048, row-major f32) into bf16 hi/lo planes,
// pre-packed into WMMA B-fragment order:
//   fragment (kt, nt) -> 32 lanes x 16 halves contiguous (1024 B)
//   lane<16: col n = nt*16+lane, K = kt*32 + {0..7, 16..23}
//   lane>=16: col n = nt*16+(lane-16), K = kt*32 + {8..15, 24..31}
// One thread per (fragment, lane): 8*128*32 = 32768 threads.
// =====================================================================
__global__ __launch_bounds__(256) void pack_w_kernel(
    const float* __restrict__ W,
    unsigned short* __restrict__ phi,
    unsigned short* __restrict__ plo) {
  int t    = blockIdx.x * blockDim.x + threadIdx.x;
  int lane = t & 31;
  int frag = t >> 5;              // frag = kt*128 + nt
  int nt   = frag & 127;
  int kt   = frag >> 7;
  int n    = nt * 16 + (lane & 15);
  int kb   = kt * 32 + ((lane >> 4) << 3);
  size_t base = ((size_t)frag * 32 + lane) * 16;
  #pragma unroll
  for (int j = 0; j < 16; ++j) {
    float w = W[(size_t)(kb + j + (j >= 8 ? 8 : 0)) * H_DIM + n];
    unsigned short h = f32_to_bf16(w);
    phi[base + j] = h;
    plo[base + j] = f32_to_bf16(w - bf16_to_f32(h));
  }
}

// =====================================================================
// Kernel 1: attention + context. One wave (32 lanes) per row.
// Lane l owns F-columns {l, l+32, ..., l+224} (8 each), keeps all 12
// leads' values in registers so x is read from HBM exactly once.
// Writes context as bf16 hi/lo planes (WMMA A source).
// =====================================================================
__global__ __launch_bounds__(256) void attn_ctx_kernel(
    const float* __restrict__ x,
    const float* __restrict__ cmask,
    const float* __restrict__ query,
    const int*   __restrict__ colidx,
    const int*   __restrict__ leadpos,
    unsigned short* __restrict__ ctxhi,
    unsigned short* __restrict__ ctxlo) {
  const int wave = threadIdx.x >> 5;
  const int lane = threadIdx.x & 31;
  const int row  = blockIdx.x * 8 + wave;
  const float* xr = x + (size_t)row * IN_DIM;

  float q[8];
  #pragma unroll
  for (int j = 0; j < 8; ++j) q[j] = query[lane + 32 * j];

  float xv[L_LEADS][8];
  float sc[L_LEADS];
  #pragma unroll
  for (int l = 0; l < L_LEADS; ++l) {
    float s = 0.0f;
    #pragma unroll
    for (int j = 0; j < 8; ++j) {
      int f = lane + 32 * j;
      int c = colidx[l * F_DIM + f];   // identity gather, honored
      float v = xr[c];
      xv[l][j] = v;
      s = fmaf(v, q[j], s);
    }
    // wave32 butterfly reduction -> all lanes get the full dot product
    #pragma unroll
    for (int off = 16; off >= 1; off >>= 1) s += __shfl_xor(s, off, 32);
    sc[l] = s;
  }

  // masks / softmax / prior blending (scalar per lane, uniform in wave)
  float am[L_LEADS];
  bool anyv = false;
  #pragma unroll
  for (int l = 0; l < L_LEADS; ++l) {
    int lp = leadpos[l];
    am[l] = cmask[(size_t)row * L_LEADS + lp];
    anyv |= (am[l] > 0.0f);
  }
  float ms[L_LEADS];
  float mx = -3.4e38f;
  #pragma unroll
  for (int l = 0; l < L_LEADS; ++l) {
    bool valid = anyv ? (am[l] > 0.0f) : true;
    ms[l] = valid ? sc[l] : NEG_BIG;
    mx = fmaxf(mx, ms[l]);
  }
  float ex[L_LEADS];
  float es = 0.0f;
  #pragma unroll
  for (int l = 0; l < L_LEADS; ++l) { ex[l] = __expf(ms[l] - mx); es += ex[l]; }
  float inv_es = 1.0f / es;
  float attn_sm[L_LEADS];
  #pragma unroll
  for (int l = 0; l < L_LEADS; ++l) attn_sm[l] = ex[l] * inv_es;

  float prior[L_LEADS];
  float psum = 0.0f;
  #pragma unroll
  for (int l = 0; l < L_LEADS; ++l) { prior[l] = fmaxf(am[l], 0.0f); psum += prior[l]; }
  float inv_ps = (psum > 0.0f) ? (1.0f / psum) : 1.0f;
  #pragma unroll
  for (int l = 0; l < L_LEADS; ++l)
    prior[l] = (psum > 0.0f) ? prior[l] * inv_ps : attn_sm[l];

  float att[L_LEADS];
  float asum = 0.0f;
  #pragma unroll
  for (int l = 0; l < L_LEADS; ++l) { att[l] = attn_sm[l] * prior[l]; asum += att[l]; }
  float inv_as = (asum > 0.0f) ? (1.0f / asum) : 1.0f;
  #pragma unroll
  for (int l = 0; l < L_LEADS; ++l)
    att[l] = (asum > 0.0f) ? att[l] * inv_as : attn_sm[l];

  // context = sum_l att[l] * per_lead[l, f]; write bf16 hi/lo planes
  #pragma unroll
  for (int j = 0; j < 8; ++j) {
    float c = 0.0f;
    #pragma unroll
    for (int l = 0; l < L_LEADS; ++l) c = fmaf(att[l], xv[l][j], c);
    unsigned short h = f32_to_bf16(c);
    size_t o = (size_t)row * F_DIM + (lane + 32 * j);
    ctxhi[o] = h;
    ctxlo[o] = f32_to_bf16(c - bf16_to_f32(h));
  }
}

// =====================================================================
// Kernel 2: out = relu(context @ W + b) via V_WMMA_F32_16X16X32_BF16
// with hi/lo error compensation: a*b ~= ah*bh + al*bh + ah*bl.
// Block = 256 threads (8 waves), block tile 64M x 64N.
// Wave w: M-subtile (w&3)*16, N-strip (w>>2)*32 (two 16-wide C tiles
// sharing the A fragments).
// =====================================================================
__global__ __launch_bounds__(256) void gemm_wmma_kernel(
    const unsigned short* __restrict__ ctxhi,
    const unsigned short* __restrict__ ctxlo,
    const unsigned short* __restrict__ wphi,
    const unsigned short* __restrict__ wplo,
    const float* __restrict__ bias,
    float* __restrict__ out) {
  const int wave = threadIdx.x >> 5;
  const int lane = threadIdx.x & 31;
  const int m0   = blockIdx.x * 64 + (wave & 3) * 16;
  const int nt0  = blockIdx.y * 4 + (wave >> 2) * 2;   // 2 N-tiles: nt0, nt0+1

  const int arow = m0 + (lane & 15);
  const int asel = (lane >> 4);                        // 0: K{0..7,16..23}, 1: K{8..15,24..31}
  const unsigned short* ah_row = ctxhi + (size_t)arow * F_DIM;
  const unsigned short* al_row = ctxlo + (size_t)arow * F_DIM;

  v8f c0 = {0.f, 0.f, 0.f, 0.f, 0.f, 0.f, 0.f, 0.f};
  v8f c1 = {0.f, 0.f, 0.f, 0.f, 0.f, 0.f, 0.f, 0.f};

  #pragma unroll 2
  for (int kt = 0; kt < 8; ++kt) {           // K = 256 in steps of 32
    int kb = kt * 32 + asel * 8;
    v16bf ahi = load_a_frag(ah_row + kb);
    v16bf alo = load_a_frag(al_row + kb);

    size_t fb0 = (((size_t)(kt * 128 + nt0) * 32) + lane) * 16;
    size_t fb1 = (((size_t)(kt * 128 + nt0 + 1) * 32) + lane) * 16;
    v16bf bhi0 = load_b_frag(wphi + fb0);
    v16bf blo0 = load_b_frag(wplo + fb0);
    v16bf bhi1 = load_b_frag(wphi + fb1);
    v16bf blo1 = load_b_frag(wplo + fb1);

    c0 = wmma_bf16(ahi, bhi0, c0);
    c0 = wmma_bf16(alo, bhi0, c0);
    c0 = wmma_bf16(ahi, blo0, c0);

    c1 = wmma_bf16(ahi, bhi1, c1);
    c1 = wmma_bf16(alo, bhi1, c1);
    c1 = wmma_bf16(ahi, blo1, c1);
  }

  // Epilogue: bias + ReLU. C layout: VGPR r -> row m0 + r + (lane>=16)*8,
  // col = ntile*16 + (lane & 15).
  const int col0 = nt0 * 16 + (lane & 15);
  const int col1 = col0 + 16;
  const float b0 = bias[col0];
  const float b1 = bias[col1];
  const int rbase = m0 + asel * 8;
  #pragma unroll
  for (int r = 0; r < 8; ++r) {
    float v0 = fmaxf(c0[r] + b0, 0.0f);
    float v1 = fmaxf(c1[r] + b1, 0.0f);
    out[(size_t)(rbase + r) * H_DIM + col0] = v0;
    out[(size_t)(rbase + r) * H_DIM + col1] = v1;
  }
}

// =====================================================================
extern "C" void kernel_launch(void* const* d_in, const int* in_sizes, int n_in,
                              void* d_out, int out_size, void* d_ws, size_t ws_size,
                              hipStream_t stream) {
  (void)in_sizes; (void)n_in; (void)out_size; (void)ws_size;
  const float* x       = (const float*)d_in[0];
  const float* cmask   = (const float*)d_in[1];
  const float* query   = (const float*)d_in[2];
  const float* W       = (const float*)d_in[3];
  const float* bias    = (const float*)d_in[4];
  const int*   colidx  = (const int*)d_in[5];
  const int*   leadpos = (const int*)d_in[6];
  float* out = (float*)d_out;

  // Workspace layout (18 MB total):
  //   ctxhi: 16384*256 bf16  (8 MB)
  //   ctxlo: 16384*256 bf16  (8 MB)
  //   wphi : 256*2048  bf16  (1 MB, fragment-packed)
  //   wplo : 256*2048  bf16  (1 MB, fragment-packed)
  unsigned short* ctxhi = (unsigned short*)d_ws;
  unsigned short* ctxlo = ctxhi + (size_t)B_ROWS * F_DIM;
  unsigned short* wphi  = ctxlo + (size_t)B_ROWS * F_DIM;
  unsigned short* wplo  = wphi  + (size_t)F_DIM * H_DIM;

  // 8 ktiles * 128 ntiles * 32 lanes = 32768 threads
  pack_w_kernel<<<128, 256, 0, stream>>>(W, wphi, wplo);

  // one wave per row, 8 waves per block
  attn_ctx_kernel<<<B_ROWS / 8, 256, 0, stream>>>(x, cmask, query, colidx, leadpos,
                                                  ctxhi, ctxlo);

  // 64x64 block tiles: (16384/64) x (2048/64)
  dim3 grid(B_ROWS / 64, H_DIM / 64);
  gemm_wmma_kernel<<<grid, 256, 0, stream>>>(ctxhi, ctxlo, wphi, wplo, bias, out);
}
